// SelfAttention_24060406792480
// MI455X (gfx1250) — matrix operand order repaired and checked by reading, not compiled, not run on hardware
//
#include <hip/hip_runtime.h>
#include <hip/hip_bf16.h>

// Shapes (fixed by the reference)
#define BB 2
#define LL 2048
#define DD 1024
#define HH 16
#define DH 64

typedef __attribute__((ext_vector_type(16))) _Float16 v16h;
typedef __attribute__((ext_vector_type(8)))  _Float16 h8;
typedef __attribute__((ext_vector_type(4)))  _Float16 h4;
typedef __attribute__((ext_vector_type(8)))  float    v8f;
typedef __attribute__((ext_vector_type(4)))  float    f4;
typedef __attribute__((ext_vector_type(4)))  int      v4i;

// ---------------------------------------------------------------------------
// CDNA5 async global->LDS staging (ASYNCcnt path), guarded so a toolchain
// without the builtins falls back to sync loads + ds_store.
// Probe-learned signature: (v4i AS1* src, v4i AS3* dst, imm offset, imm cpol)
// ---------------------------------------------------------------------------
#if __has_builtin(__builtin_amdgcn_global_load_async_to_lds_b128) && \
    __has_builtin(__builtin_amdgcn_s_wait_asynccnt)
#define USE_ASYNC_LDS 1
#else
#define USE_ASYNC_LDS 0
#endif

#if USE_ASYNC_LDS
using gptr_t = __attribute__((address_space(1))) v4i*;
using lptr_t = __attribute__((address_space(3))) v4i*;

__device__ __forceinline__ gptr_t gbl_cast(const void* p) {
  return (gptr_t)(unsigned long)p;
}
__device__ __forceinline__ lptr_t lds_cast(void* p) {
  // flat LDS address carries the LDS byte offset in its low 32 bits
  return (lptr_t)(unsigned long)(unsigned int)(unsigned long)p;
}
#define ASYNC_B128(src, dst) \
  __builtin_amdgcn_global_load_async_to_lds_b128(gbl_cast(src), lds_cast(dst), 0, 0)
#define ASYNC_WAIT() __builtin_amdgcn_s_wait_asynccnt(0)
#else
#define ASYNC_B128(src, dst) (*(h8*)(dst) = *(const h8*)(src))
#define ASYNC_WAIT() ((void)0)
#endif

__device__ __forceinline__ v8f wmma_f16(v16h a, v16h b, v8f c) {
  // D = A(16x32 f16) * B(32x16 f16) + C(16x16 f32)
  return __builtin_amdgcn_wmma_f32_16x16x32_f16(
      /*neg_a=*/false, a, /*neg_b=*/false, b,
      /*c_mod=*/(short)0, c, /*reuse_a=*/false, /*reuse_b=*/false);
}

__device__ __forceinline__ v8f vzero8() {
  v8f z;
#pragma unroll
  for (int i = 0; i < 8; ++i) z[i] = 0.0f;
  return z;
}

// ---------------------------------------------------------------------------
// Kernel 0: elementwise f32 -> f16 (vectorized x4). Converts x and the four
// weight matrices ONCE so the GEMM hot loops carry no v_cvt traffic.
// ---------------------------------------------------------------------------
__global__ __launch_bounds__(256)
void cvt_f16_kernel(const float* __restrict__ src, _Float16* __restrict__ dst,
                    int n4) {
  const int i = blockIdx.x * 256 + threadIdx.x;
  if (i < n4) {
    f4 v = ((const f4*)src)[i];
    h4 o;
#pragma unroll
    for (int j = 0; j < 4; ++j) o[j] = (_Float16)v[j];
    ((h4*)dst)[i] = o;
  }
}

// ---------------------------------------------------------------------------
// Kernel 1: QKV projection.  y[n,o] = sum_d x[n,d] * W[o,d]  (x @ W^T), f16 in.
// grid = (BL/64, D/64, 3), block = 128 (4 waves).  Wave tile: 16(M) x 64(N).
// B tile (shared by all 4 waves) staged in LDS via async global->LDS.
// N-tile == one head -> result written head-major [B,H,L,Dh] f16.
// ---------------------------------------------------------------------------
__global__ __launch_bounds__(128)
void qkv_proj_kernel(const _Float16* __restrict__ xh,
                     const _Float16* __restrict__ Wqh,
                     const _Float16* __restrict__ Wkh,
                     const _Float16* __restrict__ Wvh,
                     _Float16* __restrict__ qf,
                     _Float16* __restrict__ kf,
                     _Float16* __restrict__ vf) {
  __shared__ __align__(16) _Float16 Ws[64][40];   // W tile 64(N) x 32(K), pad 8

  const int tid  = threadIdx.x;
  const int wave = tid >> 5;
  const int lane = tid & 31;
  const int half = lane >> 4;
  const int ln   = lane & 15;

  const int z = blockIdx.z;
  const _Float16* W   = (z == 0) ? Wqh : (z == 1) ? Wkh : Wvh;
  _Float16*       dst = (z == 0) ? qf : (z == 1) ? kf : vf;

  const int m0 = blockIdx.x * 64 + wave * 16;   // row in [0, B*L)
  const int n0 = blockIdx.y * 64;               // col base; head h = blockIdx.y
  const int h  = blockIdx.y;

  const int so = tid >> 1;            // staged output row 0..63
  const int sk = (tid & 1) * 16;      // staged k segment

  v8f acc[4];
#pragma unroll
  for (int t = 0; t < 4; ++t) acc[t] = vzero8();

  for (int k0 = 0; k0 < DD; k0 += 32) {
    // ---- stage B tile: W[n0..n0+63][k0..k0+31] -> LDS (async) ----
    {
      const _Float16* src = W + (size_t)(n0 + so) * DD + k0 + sk;
      ASYNC_B128(src, &Ws[so][sk]);
      ASYNC_B128(src + 8, &Ws[so][sk + 8]);
    }
    // ---- A operand: x tile 16x32 f16, contiguous k = 16*half + i ----
    v16h a;
    {
      const int row = m0 + ln;
      const _Float16* p = xh + (size_t)row * DD + k0 + 16 * half;
      __builtin_prefetch(p + 64, 0, 0);
      h8 r0 = *(const h8*)p;
      h8 r1 = *(const h8*)(p + 8);
#pragma unroll
      for (int i = 0; i < 8; ++i) { a[i] = r0[i]; a[8 + i] = r1[i]; }
    }
    ASYNC_WAIT();
    __syncthreads();
    // ---- 4 B operands from LDS + WMMA ----
#pragma unroll
    for (int t = 0; t < 4; ++t) {
      const _Float16* p = &Ws[t * 16 + ln][16 * half];
      h8 r0 = *(const h8*)p;
      h8 r1 = *(const h8*)(p + 8);
      v16h bv;
#pragma unroll
      for (int i = 0; i < 8; ++i) { bv[i] = r0[i]; bv[8 + i] = r1[i]; }
      acc[t] = wmma_f16(a, bv, acc[t]);
    }
    __syncthreads();
  }

  // Store head-major f16: [(b*H + h)*L + l]*64 + dh
#pragma unroll
  for (int t = 0; t < 4; ++t) {
#pragma unroll
    for (int v = 0; v < 8; ++v) {
      const int row = m0 + v + 8 * half;  // = b*L + l
      const int bi  = row >> 11;          // /L
      const int l   = row & (LL - 1);
      const int dh  = t * 16 + ln;
      dst[(((size_t)(bi * HH + h)) * LL + l) * DH + dh] = (_Float16)acc[t][v];
    }
  }
}

// ---------------------------------------------------------------------------
// Kernel 2: causal flash attention with padding mask, online softmax.
// grid = B*H*(L/64) blocks of 128 threads (4 waves x 16 queries).
// Rank-1 user bias is constant along the key axis -> cancels in softmax.
// ---------------------------------------------------------------------------
__global__ __launch_bounds__(128)
void flash_attn_kernel(const _Float16* __restrict__ qg,
                       const _Float16* __restrict__ kg,
                       const _Float16* __restrict__ vg,
                       const int* __restrict__ mask,
                       _Float16* __restrict__ ctx) {
  __shared__ __align__(16) _Float16 Kc[32][72];      // K chunk, row-major
  __shared__ __align__(16) _Float16 Vt[64][40];      // V chunk, transposed
  __shared__ __align__(16) _Float16 Pb[4][16][40];   // per-wave P round-trip

  const int tid  = threadIdx.x;
  const int wave = tid >> 5;
  const int lane = tid & 31;
  const int half = lane >> 4;
  const int ln   = lane & 15;

  const int blk   = blockIdx.x;
  const int qtile = blk & 31;        // L/64 = 32
  const int bh    = blk >> 5;        // 0..B*H-1
  const int b     = bh >> 4;         // /H
  const int h     = bh & (HH - 1);

  const _Float16* qbh = qg + (size_t)bh * LL * DH;
  const _Float16* kbh = kg + (size_t)bh * LL * DH;
  const _Float16* vbh = vg + (size_t)bh * LL * DH;

  const int qb = qtile * 64;
  const int q0 = qb + wave * 16;

  // Load Q (16x64) into two A operands
  v16h aq[2];
  {
    const int row = q0 + ln;
#pragma unroll
    for (int t = 0; t < 2; ++t) {
      const _Float16* p = qbh + (size_t)row * DH + t * 32 + 16 * half;
      h8 r0 = *(const h8*)p;
      h8 r1 = *(const h8*)(p + 8);
#pragma unroll
      for (int i = 0; i < 8; ++i) { aq[t][i] = r0[i]; aq[t][8 + i] = r1[i]; }
    }
  }

  v8f acc[4];
#pragma unroll
  for (int t = 0; t < 4; ++t) acc[t] = vzero8();
  float mrow[8], lrow[8];
#pragma unroll
  for (int v = 0; v < 8; ++v) { mrow[v] = -INFINITY; lrow[v] = 0.0f; }

  const int nch = (qb >> 5) + 2;     // chunks of 32 keys up to qb+64
  const float scale = 0.125f;        // 1/sqrt(Dh)

  for (int c = 0; c < nch; ++c) {
    const int k0 = c * 32;

    // ---- cooperative staging: K row-major (async), V transposed (sync) ----
    {
      const int j   = tid >> 2;           // key 0..31
      const int seg = (tid & 3) * 16;     // dh segment
      const int gk  = k0 + j;
      const _Float16* sk = kbh + (size_t)gk * DH + seg;
      ASYNC_B128(sk, &Kc[j][seg]);
      ASYNC_B128(sk + 8, &Kc[j][seg + 8]);
      const _Float16* sv = vbh + (size_t)gk * DH + seg;
      h8 b0 = *(const h8*)sv;
      h8 b1 = *(const h8*)(sv + 8);
#pragma unroll
      for (int i = 0; i < 8; ++i) {
        Vt[seg + i][j]     = b0[i];
        Vt[seg + 8 + i][j] = b1[i];
      }
    }
    ASYNC_WAIT();
    __syncthreads();

    if (k0 <= q0 + 15) {   // causal: this chunk intersects our query tile
      // padding-mask values for our two key columns
      int mval[2];
#pragma unroll
      for (int t = 0; t < 2; ++t)
        mval[t] = mask[b * LL + k0 + t * 16 + ln];

      // ---- S = Q * K^T : two 16x16 tiles ----
      v8f s[2];
#pragma unroll
      for (int t = 0; t < 2; ++t) {
        v8f cc = vzero8();
#pragma unroll
        for (int kd = 0; kd < 2; ++kd) {
          const _Float16* p = &Kc[t * 16 + ln][kd * 32 + 16 * half];
          h8 r0 = *(const h8*)p;
          h8 r1 = *(const h8*)(p + 8);
          v16h bk;
#pragma unroll
          for (int i = 0; i < 8; ++i) { bk[i] = r0[i]; bk[8 + i] = r1[i]; }
          cc = wmma_f16(aq[kd], bk, cc);
        }
        s[t] = cc;
      }

      // ---- online softmax per row (row = v + 8*half, col = lane&15) ----
#pragma unroll
      for (int v = 0; v < 8; ++v) {
        const int gq = q0 + v + 8 * half;
#pragma unroll
        for (int t = 0; t < 2; ++t) {
          const int gk = k0 + t * 16 + ln;
          float sv = s[t][v] * scale;
          if (gk > gq || mval[t] == 0) sv = -INFINITY;
          s[t][v] = sv;
        }
        float rm = fmaxf(s[0][v], s[1][v]);
#pragma unroll
        for (int off = 1; off < 16; off <<= 1)
          rm = fmaxf(rm, __shfl_xor(rm, off, 32));
        const float mnew = fmaxf(mrow[v], rm);
        const float alpha =
            (mrow[v] <= -1e30f) ? 0.0f : __expf(mrow[v] - mnew);
        float rs = 0.0f;
#pragma unroll
        for (int t = 0; t < 2; ++t) {
          float p = (mnew <= -1e30f) ? 0.0f : __expf(s[t][v] - mnew);
          s[t][v] = p;
          rs += p;
        }
#pragma unroll
        for (int off = 1; off < 16; off <<= 1)
          rs += __shfl_xor(rs, off, 32);
        lrow[v] = lrow[v] * alpha + rs;
        mrow[v] = mnew;
#pragma unroll
        for (int t = 0; t < 4; ++t) acc[t][v] *= alpha;
        // stash P (C layout -> LDS, reread in A layout)
#pragma unroll
        for (int t = 0; t < 2; ++t)
          Pb[wave][v + 8 * half][t * 16 + ln] = (_Float16)s[t][v];
      }

      // ---- ctx += P * V : P as A operand (16x32), V tiles 32x16 ----
      v16h ap;
      {
        const _Float16* p = &Pb[wave][ln][16 * half];
        h8 r0 = *(const h8*)p;
        h8 r1 = *(const h8*)(p + 8);
#pragma unroll
        for (int i = 0; i < 8; ++i) { ap[i] = r0[i]; ap[8 + i] = r1[i]; }
      }
#pragma unroll
      for (int t = 0; t < 4; ++t) {
        const _Float16* p = &Vt[t * 16 + ln][16 * half];
        h8 r0 = *(const h8*)p;
        h8 r1 = *(const h8*)(p + 8);
        v16h bv;
#pragma unroll
        for (int i = 0; i < 8; ++i) { bv[i] = r0[i]; bv[8 + i] = r1[i]; }
        acc[t] = wmma_f16(ap, bv, acc[t]);
      }
    }
    __syncthreads();
  }

  // ---- normalize and store ctx (f16, row-major [B*L, D], col = h*64+dh) ----
  float inv[8];
#pragma unroll
  for (int v = 0; v < 8; ++v)
    inv[v] = (lrow[v] > 0.0f) ? (1.0f / lrow[v]) : 0.0f;
#pragma unroll
  for (int t = 0; t < 4; ++t) {
#pragma unroll
    for (int v = 0; v < 8; ++v) {
      const int gq = q0 + v + 8 * half;
      const int o  = h * DH + t * 16 + ln;
      ctx[((size_t)(b * LL + gq)) * DD + o] = (_Float16)(acc[t][v] * inv[v]);
    }
  }
}

// ---------------------------------------------------------------------------
// Kernel 3: output projection.  out[n,o] = sum_d ctx[n,d] * Wo[o,d]  (f32 out)
// grid = (BL/64, D/64), block = 128.  Wo (f16) staged in LDS per block.
// ---------------------------------------------------------------------------
__global__ __launch_bounds__(128)
void out_proj_kernel(const _Float16* __restrict__ ctx,
                     const _Float16* __restrict__ Woh,
                     float* __restrict__ out) {
  __shared__ __align__(16) _Float16 Ws[64][40];

  const int tid  = threadIdx.x;
  const int wave = tid >> 5;
  const int lane = tid & 31;
  const int half = lane >> 4;
  const int ln   = lane & 15;

  const int m0 = blockIdx.x * 64 + wave * 16;
  const int n0 = blockIdx.y * 64;

  const int so = tid >> 1;
  const int sk = (tid & 1) * 16;

  v8f acc[4];
#pragma unroll
  for (int t = 0; t < 4; ++t) acc[t] = vzero8();

  for (int k0 = 0; k0 < DD; k0 += 32) {
    {
      const _Float16* src = Woh + (size_t)(n0 + so) * DD + k0 + sk;
      ASYNC_B128(src, &Ws[so][sk]);
      ASYNC_B128(src + 8, &Ws[so][sk + 8]);
    }
    v16h a;
    {
      const int row = m0 + ln;
      const _Float16* p = ctx + (size_t)row * DD + k0 + 16 * half;
      __builtin_prefetch(p + 64, 0, 0);
      h8 r0 = *(const h8*)p;
      h8 r1 = *(const h8*)(p + 8);
#pragma unroll
      for (int i = 0; i < 8; ++i) { a[i] = r0[i]; a[8 + i] = r1[i]; }
    }
    ASYNC_WAIT();
    __syncthreads();
#pragma unroll
    for (int t = 0; t < 4; ++t) {
      const _Float16* p = &Ws[t * 16 + ln][16 * half];
      h8 r0 = *(const h8*)p;
      h8 r1 = *(const h8*)(p + 8);
      v16h bv;
#pragma unroll
      for (int i = 0; i < 8; ++i) { bv[i] = r0[i]; bv[8 + i] = r1[i]; }
      acc[t] = wmma_f16(a, bv, acc[t]);
    }
    __syncthreads();
  }

#pragma unroll
  for (int t = 0; t < 4; ++t) {
#pragma unroll
    for (int v = 0; v < 8; ++v) {
      const int row = m0 + v + 8 * half;
      const int o   = n0 + t * 16 + ln;
      out[(size_t)row * DD + o] = acc[t][v];
    }
  }
}

// ---------------------------------------------------------------------------
extern "C" void kernel_launch(void* const* d_in, const int* in_sizes, int n_in,
                              void* d_out, int out_size, void* d_ws,
                              size_t ws_size, hipStream_t stream) {
  (void)in_sizes; (void)n_in; (void)out_size; (void)ws_size;

  const float* x    = (const float*)d_in[0];
  const int*   mask = (const int*)d_in[1];
  // d_in[2] = user_emb: unused — the rank-1 bias is constant along the
  // softmax (key) axis, so it cancels exactly (softmax shift invariance;
  // -inf + c = -inf for masked entries).
  const float* Wq = (const float*)d_in[3];
  const float* Wk = (const float*)d_in[4];
  const float* Wv = (const float*)d_in[5];
  const float* Wo = (const float*)d_in[6];
  float* out = (float*)d_out;

  // f16 workspace layout (elements):
  //   xh [B*L*D]  | Wq/Wk/Wv/Wo [D*D] each | q/k/v [B,H,L,Dh] | ctx [B*L,D]
  const size_t NX = (size_t)BB * LL * DD;   // 4,194,304
  const size_t NW = (size_t)DD * DD;        // 1,048,576
  _Float16* xh  = (_Float16*)d_ws;
  _Float16* Wqh = xh + NX;
  _Float16* Wkh = Wqh + NW;
  _Float16* Wvh = Wkh + NW;
  _Float16* Woh = Wvh + NW;
  _Float16* qf  = Woh + NW;
  _Float16* kf  = qf + NX;
  _Float16* vf  = kf + NX;
  _Float16* cf  = vf + NX;

  // 0) one-time f32 -> f16 conversions (kills v_cvt traffic in GEMM loops)
  cvt_f16_kernel<<<dim3((int)(NX / 4 / 256)), 256, 0, stream>>>(x, xh, (int)(NX / 4));
  cvt_f16_kernel<<<dim3((int)(NW / 4 / 256)), 256, 0, stream>>>(Wq, Wqh, (int)(NW / 4));
  cvt_f16_kernel<<<dim3((int)(NW / 4 / 256)), 256, 0, stream>>>(Wk, Wkh, (int)(NW / 4));
  cvt_f16_kernel<<<dim3((int)(NW / 4 / 256)), 256, 0, stream>>>(Wv, Wvh, (int)(NW / 4));
  cvt_f16_kernel<<<dim3((int)(NW / 4 / 256)), 256, 0, stream>>>(Wo, Woh, (int)(NW / 4));

  dim3 blk(128);
  qkv_proj_kernel<<<dim3((BB * LL) / 64, DD / 64, 3), blk, 0, stream>>>(
      xh, Wqh, Wkh, Wvh, qf, kf, vf);
  flash_attn_kernel<<<dim3(BB * HH * (LL / 64)), blk, 0, stream>>>(
      qf, kf, vf, mask, cf);
  out_proj_kernel<<<dim3((BB * LL) / 64, DD / 64), blk, 0, stream>>>(
      cf, Woh, out);
}
